// MyNNUE_6262062317934
// MI455X (gfx1250) — matrix operand order repaired and checked by reading, not compiled, not run on hardware
//
#include <hip/hip_runtime.h>

// ---------------------------------------------------------------------------
// MI455X / gfx1250 NNUE evaluation.
// HBM-bound (545 MB of accum @ 23.3 TB/s ~ 23 us floor). Layer-1 GEMM
// (34 GFLOP dense over all 8 stacks) on V_WMMA_F32_16X16X32_F16 overlaps the
// stream. Packed f16 weights (272 KB) are staged once per workgroup into LDS
// by the Tensor Data Mover (tensor_load_to_lds), then all 8 waves consume
// B-fragments via ds_load_b128. One wave32 owns a 16-row tile (M=16).
// ---------------------------------------------------------------------------

typedef __attribute__((ext_vector_type(16))) _Float16 v16h;
typedef __attribute__((ext_vector_type(8)))  _Float16 v8h;
typedef __attribute__((ext_vector_type(8)))  float    v8f;
typedef __attribute__((ext_vector_type(4)))  float    v4f;
typedef __attribute__((ext_vector_type(4)))  unsigned int u32x4;
typedef __attribute__((ext_vector_type(8)))  int      i32x8;
typedef __attribute__((ext_vector_type(4)))  int      i32x4;

#define ROW_STRIDE 1040      // accum columns
#define WAVES_PER_BLOCK 8
#define W1_PACKED_HALVES (8 * 32 * 32 * 16)   // 8 Ntiles x 32 Ktiles x 32 lanes x 16 halves
#define W2_PACKED_HALVES (16 * 32 * 16)       // 16 Ntiles x 32 lanes x 16 halves (K padded to 32)
#define L1_SCRATCH_HALVES (WAVES_PER_BLOCK * 16 * 16)
#define L2_SCRATCH_HALVES (WAVES_PER_BLOCK * 16 * 32)
#define SMEM_BYTES ((W1_PACKED_HALVES + W2_PACKED_HALVES + L1_SCRATCH_HALVES + L2_SCRATCH_HALVES) * 2)

__device__ __forceinline__ float clamp127(float x) {
  return fminf(fmaxf(x, 0.0f), 127.0f);
}

// (int32)x >> 6, clipped to [0,127] -> float (matches shift_and_clamp + relu)
__device__ __forceinline__ float shift_clamp(float x) {
  int xi = ((int)x) >> 6;
  xi = xi < 0 ? 0 : (xi > 127 ? 127 : xi);
  return (float)xi;
}

// ---------------------------------------------------------------------------
// Pack W1 (128x1024 f32, row-major) and W2 (256x16 f32) into f16 B-fragments:
// element ((tn*KT + tk)*32 + lane)*16 + h holds
//   W[n = tn*16 + lane%16, k = tk*32 + (h/8)*16 + (lane/16)*8 + h%8]
// so each lane's 16 halves are contiguous (two b128 loads per fragment).
// W2 has K=16 padded to 32 with zeros (halves 8..15 of every lane are zero).
// ---------------------------------------------------------------------------
__global__ void nnue_pack_weights(const float* __restrict__ W1,
                                  const float* __restrict__ W2,
                                  _Float16* __restrict__ wsW1,
                                  _Float16* __restrict__ wsW2) {
  const int i = blockIdx.x * blockDim.x + threadIdx.x;
  if (i < W1_PACKED_HALVES) {
    const int h    = i & 15;
    const int lane = (i >> 4) & 31;
    const int tk   = (i >> 9) & 31;
    const int tn   = i >> 14;
    const int n = tn * 16 + (lane & 15);
    const int k = tk * 32 + (h >> 3) * 16 + (lane >> 4) * 8 + (h & 7);
    wsW1[i] = (_Float16)W1[n * 1024 + k];
  } else if (i < W1_PACKED_HALVES + W2_PACKED_HALVES) {
    const int j    = i - W1_PACKED_HALVES;
    const int h    = j & 15;
    const int lane = (j >> 4) & 31;
    const int tn   = j >> 9;                 // 0..15
    const int n      = tn * 16 + (lane & 15);
    const int klocal = (h >> 3) * 16 + (lane >> 4) * 8 + (h & 7); // 0..31, valid K is 0..15
    wsW2[j] = (klocal < 16) ? (_Float16)W2[n * 16 + klocal] : (_Float16)0.0f;
  }
}

// ---------------------------------------------------------------------------
// Main kernel: one wave per 16-row tile, 8 waves per block. Weights live in
// LDS, staged by the Tensor Data Mover at block start.
// ---------------------------------------------------------------------------
__global__ __launch_bounds__(256) void nnue_eval(
    const float*     __restrict__ accum,
    const int*       __restrict__ psqt_p,
    const long long* __restrict__ lsi,
    const float*     __restrict__ us,
    const float*     __restrict__ b1,
    const float*     __restrict__ b2,
    const float*     __restrict__ Wout,
    const float*     __restrict__ bout,
    const _Float16*  __restrict__ wsW1,    // packed weights in workspace (W2 contiguous after W1)
    float*           __restrict__ out,
    int nrows) {
  extern __shared__ __align__(16) char smem[];
  _Float16* lds_w1 = (_Float16*)smem;                       // W1 fragments
  _Float16* lds_w2 = lds_w1 + W1_PACKED_HALVES;             // W2 fragments
  _Float16* lds_l1 = lds_w2 + W2_PACKED_HALVES;             // per-wave 16x16 l1c
  _Float16* lds_l2 = lds_l1 + L1_SCRATCH_HALVES;            // per-wave 16x32 l2c

  const int lane = threadIdx.x & 31;
  const int wave = threadIdx.x >> 5;

  // -------- Stage all packed weights (272 KB) into LDS via the TDM ---------
  if (wave == 0) {
    const unsigned long long ga = (unsigned long long)(uintptr_t)wsW1;
    const unsigned lds_base = (unsigned)(uintptr_t)(void*)lds_w1;  // LDS byte offset
    const unsigned elems = (W1_PACKED_HALVES + W2_PACKED_HALVES) / 4;  // 8-byte elements
    u32x4 g0;
    g0[0] = 1u;                                              // count=1 (valid user D#)
    g0[1] = lds_base;                                        // D#.lds_addr
    g0[2] = (unsigned)ga;                                    // global_addr[31:0]
    g0[3] = ((unsigned)(ga >> 32) & 0x01FFFFFFu) | 0x80000000u; // addr[56:32] | type=2
    i32x8 g1 = {};
    g1[0] = (int)(3u << 16);                                 // data_size = 8 bytes
    g1[1] = (int)((elems & 0xFFFFu) << 16);                  // tensor_dim0[15:0]
    g1[2] = (int)(((elems >> 16) & 0xFFFFu) | (1u << 16));   // tensor_dim0[31:16] | tensor_dim1=1
    g1[3] = (int)((elems & 0xFFFFu) << 16);                  // tile_dim0 (1-D tile)
    g1[4] = 0;                                               // tile_dim1/2 unused
    g1[5] = (int)elems;                                      // tensor_dim0_stride[31:0]
    g1[6] = 0;
    g1[7] = 0;
    const i32x4 gz = {};
    const i32x8 gz8 = {};
    __builtin_amdgcn_tensor_load_to_lds(g0, g1, gz, gz, gz8, 0);
    __builtin_amdgcn_s_wait_tensorcnt(0);
  }
  __syncthreads();

  const int row0 = (blockIdx.x * WAVES_PER_BLOCK + wave) * 16;
  if (row0 < nrows) {
    const int m      = lane & 15;   // row for A-frag / column N for B,C-frags
    const int halfid = lane >> 4;
    const float* __restrict__ arow = accum + (size_t)(row0 + m) * ROW_STRIDE;

    // lsi for the rows this lane's C-fragment elements belong to (M = r + 8*halfid)
    int lsiv[8];
#pragma unroll
    for (int r = 0; r < 8; ++r)
      lsiv[r] = (int)lsi[row0 + r + 8 * halfid];

    // ---------------- Layer 1: M=16, K=1024, N=128 (8 N-tiles) -------------
    v8f acc[8] = {};
    const v16h* __restrict__ w1l = (const v16h*)lds_w1;

    for (int tk = 0; tk < 32; ++tk) {
      // A fragment (16-bit A layout): lane-half holds K runs [halfid*8,+8)
      // and [16+halfid*8,+8) of the K-tile -> two contiguous 8-float runs in
      // accum (main col c maps to accum col c<512 ? c : c+8).
      const int kb0 = tk * 32 + halfid * 8;
      const int kb1 = kb0 + 16;
      const int g0c = (kb0 < 512) ? kb0 : kb0 + 8;
      const int g1c = (kb1 < 512) ? kb1 : kb1 + 8;
      const v4f a0 = *(const v4f*)(arow + g0c);
      const v4f a1 = *(const v4f*)(arow + g0c + 4);
      const v4f a2 = *(const v4f*)(arow + g1c);
      const v4f a3 = *(const v4f*)(arow + g1c + 4);

      if (tk + 1 < 32) {  // stream next K-tile of accum (global_prefetch_b8)
        const int nb = (tk + 1) * 32 + halfid * 8;
        __builtin_prefetch(arow + ((nb < 512) ? nb : nb + 8), 0, 1);
      }

      v16h A;
#pragma unroll
      for (int j = 0; j < 4; ++j) {
        A[j]      = (_Float16)clamp127(a0[j]);
        A[4 + j]  = (_Float16)clamp127(a1[j]);
        A[8 + j]  = (_Float16)clamp127(a2[j]);
        A[12 + j] = (_Float16)clamp127(a3[j]);
      }

      const int bbase = tk * 32 + lane;
#pragma unroll
      for (int tn = 0; tn < 8; ++tn) {
        const v16h Bv = w1l[tn * 1024 + bbase];   // ds_load_b128 x2 from LDS
        acc[tn] = __builtin_amdgcn_wmma_f32_16x16x32_f16(
            false, A, false, Bv, (short)0, acc[tn], false, false);
      }
    }

    // bias, per-row stack selection on C layout, quantize, stash to LDS
#pragma unroll
    for (int t = 0; t < 8; ++t) {
      const float bb = b1[t * 16 + m];
#pragma unroll
      for (int r = 0; r < 8; ++r) acc[t][r] += bb;
    }
#pragma unroll
    for (int r = 0; r < 8; ++r) {
      const int t = lsiv[r];
      float x = acc[0][r];
#pragma unroll
      for (int tt = 1; tt < 8; ++tt) x = (t == tt) ? acc[tt][r] : x;
      lds_l1[wave * 256 + (r + 8 * halfid) * 16 + m] = (_Float16)shift_clamp(x);
    }
    asm volatile("s_wait_dscnt 0" ::: "memory");  // wave-local LDS transpose

    // ---------------- Layer 2: M=16, K=16 (padded to 32), N=256 ------------
    const v8h a2lo = *(const v8h*)&lds_l1[wave * 256 + m * 16 + halfid * 8];
    v16h A2;
#pragma unroll
    for (int j = 0; j < 8; ++j) {
      A2[j]     = a2lo[j];
      A2[8 + j] = (_Float16)0.0f;   // K=16..31 padding (weights also zero there)
    }

    v8f acc2[16] = {};
    const v16h* __restrict__ w2l = (const v16h*)lds_w2;
#pragma unroll
    for (int tn = 0; tn < 16; ++tn) {
      const v16h Bv = w2l[tn * 32 + lane];
      acc2[tn] = __builtin_amdgcn_wmma_f32_16x16x32_f16(
          false, A2, false, Bv, (short)0, acc2[tn], false, false);
    }
#pragma unroll
    for (int t = 0; t < 16; ++t) {
      const float bb = b2[t * 16 + m];
#pragma unroll
      for (int r = 0; r < 8; ++r) acc2[t][r] += bb;
    }
#pragma unroll
    for (int r = 0; r < 8; ++r) {
      const int s = lsiv[r];
#pragma unroll
      for (int u = 0; u < 2; ++u) {
        const int t = 2 * s + u;  // stack s covers N-tiles 2s, 2s+1
        float x = acc2[0][r];
#pragma unroll
        for (int tt = 1; tt < 16; ++tt) x = (t == tt) ? acc2[tt][r] : x;
        lds_l2[wave * 512 + (r + 8 * halfid) * 32 + u * 16 + m] =
            (_Float16)shift_clamp(x);
      }
    }
    asm volatile("s_wait_dscnt 0" ::: "memory");

    // ---------------- Layer 3 (8x32) + PSQT, lanes 0..15 = rows ------------
    if (lane < 16) {
      const int row = row0 + lane;
      const int q   = (int)lsi[row];
      const float* __restrict__ wq = Wout + q * 32;
      float s = bout[q];
#pragma unroll
      for (int k = 0; k < 32; ++k)
        s += (float)lds_l2[wave * 512 + lane * 32 + k] * wq[k];
      const int p  = *psqt_p;
      const float wp = accum[(size_t)row * ROW_STRIDE + 512 + p];
      const float bp = accum[(size_t)row * ROW_STRIDE + 1032 + p];
      out[row] = s + (wp - bp) * (us[row] - 0.5f);
    }
  }
}

// ---------------------------------------------------------------------------
extern "C" void kernel_launch(void* const* d_in, const int* in_sizes, int n_in,
                              void* d_out, int out_size, void* d_ws, size_t ws_size,
                              hipStream_t stream) {
  const float*     accum = (const float*)d_in[0];
  const int*       psqt  = (const int*)d_in[1];
  const long long* lsi   = (const long long*)d_in[2];   // jnp.int64
  const float*     us    = (const float*)d_in[3];
  const float*     W1    = (const float*)d_in[4];
  const float*     b1    = (const float*)d_in[5];
  const float*     W2    = (const float*)d_in[6];
  const float*     b2    = (const float*)d_in[7];
  const float*     Wout  = (const float*)d_in[8];
  const float*     bout  = (const float*)d_in[9];
  float*           out   = (float*)d_out;
  const int nrows = in_sizes[2];   // B (size of layer_stack_indices)

  _Float16* wsW1 = (_Float16*)d_ws;
  _Float16* wsW2 = wsW1 + W1_PACKED_HALVES;

  const int pack_elems = W1_PACKED_HALVES + W2_PACKED_HALVES;
  nnue_pack_weights<<<(pack_elems + 255) / 256, 256, 0, stream>>>(W1, W2, wsW1, wsW2);

  static bool attr_set = false;   // idempotent, deterministic; not a stream op
  if (!attr_set) {
    (void)hipFuncSetAttribute(reinterpret_cast<const void*>(nnue_eval),
                              hipFuncAttributeMaxDynamicSharedMemorySize, SMEM_BYTES);
    attr_set = true;
  }

  const int tiles  = (nrows + 15) / 16;
  const int blocks = (tiles + WAVES_PER_BLOCK - 1) / WAVES_PER_BLOCK;
  nnue_eval<<<blocks, 256, SMEM_BYTES, stream>>>(accum, psqt, lsi, us, b1, b2,
                                                 Wout, bout, wsW1, out, nrows);
}
